// SiameseClassifier_68805376082289
// MI455X (gfx1250) — matrix operand order repaired and checked by reading, not compiled, not run on hardware
//
#include <hip/hip_runtime.h>
#include <math.h>

// ---------------------------------------------------------------------------
// Siamese LSTM (V=32000, E=1024, H=1024, T=128, B=128) for gfx1250 (MI455X).
//   prep:  wcat[4096][2048] bf16 = [w_ih | w_hh], bias = b_ih + b_hh,
//          emb -> bf16 copy, h0/c0 -> state buffers (h kept in bf16 + f32).
//   step kernel (x128): gates = [x_t ; h] @ wcat^T via v_wmma_f32_16x16x32_bf16.
//          A tiles staged with global_load_async_to_lds_b128 (ASYNCcnt),
//          double-buffered in LDS so chunk i+1 copies overlap chunk i WMMAs.
//          LSTM cell finished in-block (gate-interleaved N mapping).
//   final: out[m] = exp(-sum |h_a - h_b|).
// ---------------------------------------------------------------------------

typedef __bf16 bf16;
typedef __attribute__((ext_vector_type(16))) __bf16 bf16x16;
typedef __attribute__((ext_vector_type(8)))  __bf16 bf16x8;
typedef __attribute__((ext_vector_type(8)))  float  f32x8;

#define VOCAB 32000
#define EDIM  1024
#define HDIM  1024
#define TSEQ  128
#define BATCH 128
#define KTOT  2048            // E + H concatenated
#define NTOT  4096            // 4 gates * H

// step-kernel tiling
#define BM       64           // batch rows per block
#define KC       64           // K chunk staged in LDS
#define NCHUNK   (KTOT / KC)  // 32
#define A_STRIDE (KC + 8)     // bf16 elems, pad to dodge bank conflicts
#define G_STRIDE (128 + 8)    // fp32 gate-tile stride in LDS

__device__ __forceinline__ float sigmoidf_(float x) {
    return 1.0f / (1.0f + expf(-x));
}

// 32 bytes global->LDS via the async data mover (ASYNCcnt path).
// INST_OFFSET is added to BOTH the LDS and the global address (ISA 08 §4.4).
__device__ __forceinline__ void async_copy32(unsigned lds_off, const bf16* src) {
    asm volatile(
        "global_load_async_to_lds_b128 %0, %1, off\n\t"
        "global_load_async_to_lds_b128 %0, %1, off offset:16"
        :: "v"(lds_off), "v"((unsigned long long)(uintptr_t)src)
        : "memory");
}

// ---------------------------------------------------------------------------
// Prep kernels
// ---------------------------------------------------------------------------
__global__ __launch_bounds__(256) void prep_weights_kernel(
    const float* __restrict__ w_ih, const float* __restrict__ w_hh,
    const float* __restrict__ b_ih, const float* __restrict__ b_hh,
    bf16* __restrict__ wcat, float* __restrict__ bias)
{
    const size_t total  = (size_t)NTOT * KTOT;
    const size_t stride = (size_t)gridDim.x * blockDim.x;
    size_t idx = (size_t)blockIdx.x * blockDim.x + threadIdx.x;
    for (size_t i = idx; i < total; i += stride) {
        size_t n = i >> 11;          // / 2048
        size_t k = i & 2047;
        float v = (k < EDIM) ? w_ih[n * EDIM + k]
                             : w_hh[n * HDIM + (k - EDIM)];
        wcat[i] = (bf16)v;
    }
    if (idx < NTOT) bias[idx] = b_ih[idx] + b_hh[idx];
}

__global__ __launch_bounds__(256) void prep_emb_kernel(
    const float* __restrict__ emb, bf16* __restrict__ emb_bf)
{
    const size_t total  = (size_t)VOCAB * EDIM;
    const size_t stride = (size_t)gridDim.x * blockDim.x;
    for (size_t i = (size_t)blockIdx.x * blockDim.x + threadIdx.x;
         i < total; i += stride)
        emb_bf[i] = (bf16)emb[i];
}

__global__ __launch_bounds__(256) void prep_state_kernel(
    const float* __restrict__ h0a, const float* __restrict__ c0a,
    const float* __restrict__ h0b, const float* __restrict__ c0b,
    bf16* __restrict__ habf, float* __restrict__ ca,
    bf16* __restrict__ hbbf, float* __restrict__ cb)
{
    int i = blockIdx.x * blockDim.x + threadIdx.x;
    if (i < BATCH * HDIM) {
        habf[i] = (bf16)h0a[i]; ca[i] = c0a[i];
        hbbf[i] = (bf16)h0b[i]; cb[i] = c0b[i];
    }
}

// ---------------------------------------------------------------------------
// One LSTM timestep, both encoders. gridDim.x = 128:
// [enc(2)] x [m-tile(2)] x [col-block(32)].
// Block: 64 batch rows x (32 hidden cols x 4 gates) + cell epilogue.
// ---------------------------------------------------------------------------
__global__ __launch_bounds__(256) void lstm_step_kernel(
    const int*   __restrict__ ids_a, const int* __restrict__ ids_b,
    const bf16*  __restrict__ emb_bf,
    const bf16*  __restrict__ wcat,  const float* __restrict__ bias,
    const bf16*  __restrict__ hpbf_a, bf16* __restrict__ hnbf_a,
    float* __restrict__ hf_a, float* __restrict__ c_a,
    const bf16*  __restrict__ hpbf_b, bf16* __restrict__ hnbf_b,
    float* __restrict__ hf_b, float* __restrict__ c_b,
    int t)
{
    __shared__ bf16  Abuf[2][BM * A_STRIDE];  // double-buffered [x_t ; h] chunk
    __shared__ float Gbuf[BM * G_STRIDE];     // gate tile (fp32)

    const int bid  = blockIdx.x;
    const int enc  = bid >> 6;
    const int rem  = bid & 63;
    const int m0   = (rem >> 5) * BM;         // batch-row base
    const int col0 = (rem & 31) * 32;         // hidden-col base

    const int*  ids  = enc ? ids_b  : ids_a;
    const bf16* hpbf = enc ? hpbf_b : hpbf_a;
    bf16*       hnbf = enc ? hnbf_b : hnbf_a;
    float*      hf   = enc ? hf_b   : hf_a;
    float*      cbuf = enc ? c_b    : c_a;

    const int tid  = threadIdx.x;
    const int lane = tid & 31;
    const int wave = tid >> 5;
    const int wm   = wave & 1;                // M half (32 rows)
    const int wg   = wave >> 1;               // gate index 0..3
    const int l15  = lane & 15;
    const int half = lane >> 4;

    // ---- per-thread staging setup (once, outside the K loop) -------------
    const int r    = tid >> 2;                // local row 0..63
    const int seg  = (tid & 3) * 16;          // 32B segment within 64-K chunk
    const int m    = m0 + r;
    const int token = ids[t * BATCH + m];
    const bf16* xrow = emb_bf + (size_t)token * EDIM + seg;
    const bf16* hrow = hpbf   + (size_t)m    * HDIM + seg;
    const unsigned lds0 =
        (unsigned)(uintptr_t)&Abuf[0][r * A_STRIDE + seg];
    const unsigned lds1 =
        (unsigned)(uintptr_t)&Abuf[1][r * A_STRIDE + seg];

    f32x8 acc[2][2] = {};                     // [mf][nf] 16x16 fp32 fragments

    // prefetch chunk 0
    async_copy32(lds0, xrow);

    for (int i = 0; i < NCHUNK; ++i) {
        const int cur = i & 1;
        const int kc  = i * KC;
        // issue chunk i+1 into the other buffer, then wait for chunk i only
        if (i + 1 < NCHUNK) {
            const int kg = kc + KC;
            const bf16* s = (kg < EDIM) ? (xrow + kg) : (hrow + (kg - EDIM));
            async_copy32(cur ? lds0 : lds1, s);
            asm volatile("s_wait_asynccnt 0x2" ::: "memory");
        } else {
            asm volatile("s_wait_asynccnt 0x0" ::: "memory");
        }
        __syncthreads();                      // chunk i visible to all waves

        const bf16* Ab = &Abuf[cur][0];
#pragma unroll
        for (int kf = 0; kf < 2; ++kf) {
            // A fragments (16x32 bf16): lanes 0-15 rows M, K 0-15;
            // lanes 16-31 same rows, K 16-31.
            bf16x16 afr[2];
#pragma unroll
            for (int mf = 0; mf < 2; ++mf) {
                const int arow = wm * 32 + mf * 16 + l15;
                const int ka   = kf * 32 + half * 16;
                union { bf16x16 v; bf16x8 h[2]; } u;
                u.h[0] = *(const bf16x8*)&Ab[arow * A_STRIDE + ka];
                u.h[1] = *(const bf16x8*)&Ab[arow * A_STRIDE + ka + 8];
                afr[mf] = u.v;
            }
#pragma unroll
            for (int nf = 0; nf < 2; ++nf) {
                // B fragment (32x16 bf16): lane holds column N=l15,
                // 16 consecutive K per half-wave; row of wcat == column of B.
                const size_t brow = (size_t)(wg * HDIM + col0 + nf * 16 + l15);
                const bf16* bp = wcat + brow * KTOT + kc + kf * 32 + half * 16;
                const bf16x16 bfr = *(const bf16x16*)bp;   // 32B aligned
#pragma unroll
                for (int mf = 0; mf < 2; ++mf) {
                    acc[mf][nf] = __builtin_amdgcn_wmma_f32_16x16x32_bf16(
                        false, afr[mf], false, bfr,
                        (short)0, acc[mf][nf], false, false);
                }
            }
        }
        __syncthreads();                      // all waves done reading buf[cur]
    }

    // ---- spill gate accumulators to LDS ---------------------------------
    // C/D layout: VGPR v -> M = fragBase + (lane>>4)*8 + v, N = lane&15.
#pragma unroll
    for (int mf = 0; mf < 2; ++mf) {
#pragma unroll
        for (int nf = 0; nf < 2; ++nf) {
            const int colb = wg * 32 + nf * 16 + l15;
            const int rowb = wm * 32 + mf * 16 + half * 8;
#pragma unroll
            for (int v = 0; v < 8; ++v)
                Gbuf[(rowb + v) * G_STRIDE + colb] = acc[mf][nf][v];
        }
    }
    __syncthreads();

    // ---- LSTM cell: 64 rows x 32 cols, 8 per thread ---------------------
    {
        const int col = tid & 31;
        const int mb  = (tid >> 5) * 8;
        const float bi = bias[0 * HDIM + col0 + col];
        const float bf = bias[1 * HDIM + col0 + col];
        const float bg = bias[2 * HDIM + col0 + col];
        const float bo = bias[3 * HDIM + col0 + col];
#pragma unroll
        for (int j = 0; j < 8; ++j) {
            const int mm = mb + j;
            const float gi = Gbuf[mm * G_STRIDE + 0 * 32 + col] + bi;
            const float gf = Gbuf[mm * G_STRIDE + 1 * 32 + col] + bf;
            const float gg = Gbuf[mm * G_STRIDE + 2 * 32 + col] + bg;
            const float go = Gbuf[mm * G_STRIDE + 3 * 32 + col] + bo;
            const size_t gidx = (size_t)(m0 + mm) * HDIM + col0 + col;
            const float cp = cbuf[gidx];
            const float cn = sigmoidf_(gf) * cp + sigmoidf_(gi) * tanhf(gg);
            const float hn = sigmoidf_(go) * tanhf(cn);
            cbuf[gidx] = cn;
            hnbf[gidx] = (bf16)hn;   // bf16 h for the next step's WMMA A tile
            hf[gidx]   = hn;         // fp32 h for the final similarity
        }
    }
}

// ---------------------------------------------------------------------------
// out[m] = exp(-sum_k |h_a[m,k] - h_b[m,k]|)
// ---------------------------------------------------------------------------
__global__ __launch_bounds__(256) void similarity_kernel(
    const float* __restrict__ ha, const float* __restrict__ hb,
    float* __restrict__ out)
{
    const int m   = blockIdx.x;
    const int tid = threadIdx.x;
    float s = 0.0f;
    for (int k = tid; k < HDIM; k += 256)
        s += fabsf(ha[(size_t)m * HDIM + k] - hb[(size_t)m * HDIM + k]);
    __shared__ float red[256];
    red[tid] = s;
    __syncthreads();
    for (int off = 128; off > 0; off >>= 1) {
        if (tid < off) red[tid] += red[tid + off];
        __syncthreads();
    }
    if (tid == 0) out[m] = expf(-red[0]);
}

// ---------------------------------------------------------------------------
// Launch: prep -> 128 fused steps (both encoders) -> similarity
// ---------------------------------------------------------------------------
extern "C" void kernel_launch(void* const* d_in, const int* in_sizes, int n_in,
                              void* d_out, int out_size, void* d_ws, size_t ws_size,
                              hipStream_t stream)
{
    const int*   ids_a = (const int*)  d_in[0];
    const int*   ids_b = (const int*)  d_in[1];
    const float* h0_a  = (const float*)d_in[2];
    const float* c0_a  = (const float*)d_in[3];
    const float* h0_b  = (const float*)d_in[4];
    const float* c0_b  = (const float*)d_in[5];
    const float* emb   = (const float*)d_in[6];
    const float* w_ih  = (const float*)d_in[7];
    const float* w_hh  = (const float*)d_in[8];
    const float* b_ih  = (const float*)d_in[9];
    const float* b_hh  = (const float*)d_in[10];
    float* out = (float*)d_out;

    // workspace carve-up (all chunk sizes multiples of 256B)
    char* p = (char*)d_ws;
    bf16*  wcat   = (bf16*)p;  p += (size_t)NTOT * KTOT * sizeof(bf16);   // 16 MB
    bf16*  emb_bf = (bf16*)p;  p += (size_t)VOCAB * EDIM * sizeof(bf16);  // 62.5 MB
    float* bias   = (float*)p; p += (size_t)NTOT * sizeof(float);         // 16 KB
    const size_t stf = (size_t)BATCH * HDIM * sizeof(float);              // 512 KB
    const size_t stb = (size_t)BATCH * HDIM * sizeof(bf16);               // 256 KB
    bf16*  habf0 = (bf16*)p;  p += stb;
    bf16*  habf1 = (bf16*)p;  p += stb;
    bf16*  hbbf0 = (bf16*)p;  p += stb;
    bf16*  hbbf1 = (bf16*)p;  p += stb;
    float* hfa   = (float*)p; p += stf;
    float* hfb   = (float*)p; p += stf;
    float* ca    = (float*)p; p += stf;
    float* cb    = (float*)p; p += stf;

    prep_weights_kernel<<<4096, 256, 0, stream>>>(w_ih, w_hh, b_ih, b_hh, wcat, bias);
    prep_emb_kernel<<<8192, 256, 0, stream>>>(emb, emb_bf);
    prep_state_kernel<<<(BATCH * HDIM + 255) / 256, 256, 0, stream>>>(
        h0_a, c0_a, h0_b, c0_b, habf0, ca, hbbf0, cb);

    for (int t = 0; t < TSEQ; ++t) {
        const bf16* hpa = (t & 1) ? habf1 : habf0;
        bf16*       hna = (t & 1) ? habf0 : habf1;
        const bf16* hpb = (t & 1) ? hbbf1 : hbbf0;
        bf16*       hnb = (t & 1) ? hbbf0 : hbbf1;
        lstm_step_kernel<<<128, 256, 0, stream>>>(
            ids_a, ids_b, emb_bf, wcat, bias,
            hpa, hna, hfa, ca,
            hpb, hnb, hfb, cb, t);
    }
    similarity_kernel<<<BATCH, 256, 0, stream>>>(hfa, hfb, out);
}
